// BbBartAttention_19963007992437
// MI455X (gfx1250) — matrix operand order repaired and checked by reading, not compiled, not run on hardware
//
#include <hip/hip_runtime.h>
#include <hip/hip_bf16.h>

// Problem constants (match reference)
#define EMBED 768
#define SEQ   1024
#define BATCH 4
#define HEADS 12
#define HDIM  64
#define MROWS (BATCH * SEQ)        // 4096
#define SCALING 0.125f             // 64^-0.5

typedef unsigned int u32;
typedef __attribute__((ext_vector_type(16))) __bf16 v16bf;
typedef __attribute__((ext_vector_type(8)))  __bf16 v8bf;
typedef __attribute__((ext_vector_type(8)))  float  v8f;
typedef __attribute__((ext_vector_type(4)))  u32    u32x4;
typedef __attribute__((ext_vector_type(8)))  int    i32x8;
typedef __attribute__((ext_vector_type(4)))  int    i32x4;

// A/B fragment: 16 bf16 per lane. h[0] = elements 0..7, h[1] = elements 8..15.
union AFrag { v16bf v; v8bf h[2]; };

__device__ __forceinline__ __bf16 f2bf(float f) {
    // round-to-nearest-even f32 -> bf16
    union { unsigned short u; __bf16 b; } c;
    unsigned x = __float_as_uint(f);
    c.u = (unsigned short)((x + 0x7FFFu + ((x >> 16) & 1u)) >> 16);
    return c.b;
}

// ---------------------------------------------------------------------------
// TDM: issue a 2D tile load (global -> LDS) via Tensor Data Mover.
// D# layout per CDNA5 ISA. data_size = 2 bytes (bf16).
// LDS padding: 4 dwords inserted every 32 dwords (one 64-element row), so the
// effective LDS row stride is 144 bytes (bank-conflict spreading).
// This toolchain exposes the 6-arg builtin:
//   (u32x4 g0, i32x8 g1, i32x4 g2, i32x4 g3, i32x8 extra, i32 cpol)
// ---------------------------------------------------------------------------
#define KC 64                                  // k-extent per stage (elements)
#define LDS_ROW_B 144                          // 64*2 bytes + 16B pad
#define A_TILE_BYTES (64 * LDS_ROW_B)          // 9216
#define B_TILE_BYTES (128 * LDS_ROW_B)         // 18432
#define STAGE_BYTES (A_TILE_BYTES + B_TILE_BYTES)

__device__ __forceinline__ void tdm_load_2d(const void* gaddr, u32 lds_off,
                                            u32 tile0, u32 tile1,
                                            u32 tdim0, u32 tdim1, u32 stride0) {
    unsigned long long ga = (unsigned long long)gaddr;
    u32x4 g0;
    g0[0] = 1u;                                  // count=1, user descriptor
    g0[1] = lds_off;                             // lds_addr (bytes)
    g0[2] = (u32)ga;                             // global_addr[31:0]
    g0[3] = (u32)(ga >> 32) | (2u << 30);        // global_addr[56:32] | type=2
    i32x8 g1;
    g1[0] = (int)((1u << 16)                     // data_size = 1 -> 2 bytes
                | (1u << 20)                     // pad_enable
                | (4u << 22)                     // pad_interval: 32 dwords
                | (3u << 25));                   // pad_amount: 4 dwords
    g1[1] = (int)(tdim0 << 16);                  // tensor_dim0 [63:48]
    g1[2] = (int)((tdim0 >> 16) | (tdim1 << 16));// tensor_dim0 hi | dim1 lo
    g1[3] = (int)((tdim1 >> 16) | (tile0 << 16));// tensor_dim1 hi | tile_dim0
    g1[4] = (int)(tile1 & 0xFFFFu);              // tile_dim1 (tile_dim2 = 0)
    g1[5] = (int)stride0;                        // tensor_dim0_stride lo32
    g1[6] = 0;
    g1[7] = 0;
    i32x4 z4 = {0, 0, 0, 0};
    i32x8 z8 = {0, 0, 0, 0, 0, 0, 0, 0};
    __builtin_amdgcn_tensor_load_to_lds(g0, g1, z4, z4, z8, 0);
}

// ---------------------------------------------------------------------------
// f32 -> bf16 conversion (grid-stride)
// ---------------------------------------------------------------------------
__global__ void cvt_f32_to_bf16(const float* __restrict__ in,
                                __bf16* __restrict__ out, int n) {
    for (int i = blockIdx.x * blockDim.x + threadIdx.x; i < n;
         i += gridDim.x * blockDim.x)
        out[i] = f2bf(in[i]);
}

// ---------------------------------------------------------------------------
// WMMA GEMM with TDM double-buffered LDS staging:
//   C[m,n] = (sum_k A[m,k] * W[n,k] + bias[n]) * scale
//   A: [MROWS, EMBED] bf16 row-major     W: [EMBED, EMBED] bf16 row-major
// Epilogue modes:
//   0: f32 row-major [MROWS, EMBED] (final projection -> d_out)
//   1/2: bf16 scattered to [B, H, S, D]   (Q / K)
//   3:   bf16 scattered to [B, H, D, S]   (V transposed for PV wmma)
// Block: 256 threads = 8 waves (2m x 4n), wave tile 32x32, block tile 64x128.
// Wave 0 drives the TDM; barriers hand tiles to the compute waves.
// ---------------------------------------------------------------------------
__global__ __launch_bounds__(256)
void gemm_bf16_wmma(const __bf16* __restrict__ A, const __bf16* __restrict__ W,
                    const float* __restrict__ bias, float scale, int mode,
                    float* __restrict__ outF, __bf16* __restrict__ outB) {
    __shared__ char lds[2 * STAGE_BYTES];        // 54 KB double buffer

    const int lane = threadIdx.x & 31;
    const int wave = threadIdx.x >> 5;
    const int nn = lane & 15, hi = lane >> 4;
    const int wm = wave & 1, wn = wave >> 1;
    const int mBlock = blockIdx.x * 64;
    const int nBlock = blockIdx.y * 128;

    const u32 ldsBase = (u32)(unsigned long long)(char*)lds;  // LDS byte offset

    // prologue: stage 0
    if (wave == 0) {
        tdm_load_2d(A + (size_t)mBlock * EMBED, ldsBase,
                    KC, 64, EMBED, MROWS - mBlock, EMBED);
        tdm_load_2d(W + (size_t)nBlock * EMBED, ldsBase + A_TILE_BYTES,
                    KC, 128, EMBED, EMBED - nBlock, EMBED);
    }

    v8f acc[2][2] = {};
    int buf = 0;

    for (int k0 = 0; k0 < EMBED; k0 += KC) {
        if (wave == 0) {
            if (k0 + KC < EMBED) {
                u32 nb = ldsBase + (buf ^ 1) * STAGE_BYTES;
                tdm_load_2d(A + (size_t)mBlock * EMBED + k0 + KC, nb,
                            KC, 64, EMBED - k0 - KC, MROWS - mBlock, EMBED);
                tdm_load_2d(W + (size_t)nBlock * EMBED + k0 + KC, nb + A_TILE_BYTES,
                            KC, 128, EMBED - k0 - KC, EMBED - nBlock, EMBED);
                __builtin_amdgcn_s_wait_tensorcnt(2);  // current stage retired
            } else {
                __builtin_amdgcn_s_wait_tensorcnt(0);
            }
        }
        __syncthreads();

        const char* sA = lds + buf * STAGE_BYTES;
        const char* sB = sA + A_TILE_BYTES;

#pragma unroll
        for (int ks = 0; ks < 2; ++ks) {         // two k-substeps of 32
            AFrag a[2], b[2];
#pragma unroll
            for (int i = 0; i < 2; ++i) {
                // A fragment: lane = row; k = ks*32 + hi*16 + {0..15}
                const char* pa = sA + (wm * 32 + i * 16 + nn) * LDS_ROW_B
                               + (ks * 32 + hi * 16) * 2;
                a[i].h[0] = *(const v8bf*)pa;
                a[i].h[1] = *(const v8bf*)(pa + 16);
                // B fragment: lane = col; k = ks*32 + hi*16 + {0..15}
                const char* pb = sB + (wn * 32 + i * 16 + nn) * LDS_ROW_B
                               + (ks * 32 + hi * 16) * 2;
                b[i].h[0] = *(const v8bf*)pb;
                b[i].h[1] = *(const v8bf*)(pb + 16);
            }
#pragma unroll
            for (int i = 0; i < 2; ++i)
#pragma unroll
                for (int j = 0; j < 2; ++j)
                    acc[i][j] = __builtin_amdgcn_wmma_f32_16x16x32_bf16(
                        false, a[i].v, false, b[j].v, (short)0, acc[i][j],
                        false, false);
        }
        __syncthreads();   // all reads done before TDM overwrites this buffer
        buf ^= 1;
    }

#pragma unroll
    for (int i = 0; i < 2; ++i)
#pragma unroll
        for (int j = 0; j < 2; ++j)
#pragma unroll
            for (int r = 0; r < 8; ++r) {
                int m = mBlock + wm * 32 + i * 16 + r + 8 * hi;
                int n = nBlock + wn * 32 + j * 16 + nn;
                float x = (acc[i][j][r] + bias[n]) * scale;
                if (mode == 0) {
                    outF[(size_t)m * EMBED + n] = x;
                } else {
                    int bb = m >> 10, s = m & (SEQ - 1);
                    int h = n >> 6, d = n & (HDIM - 1);
                    if (mode == 3)
                        outB[(((size_t)bb * HEADS + h) * HDIM + d) * SEQ + s] = f2bf(x);
                    else
                        outB[(((size_t)bb * HEADS + h) * SEQ + s) * HDIM + d] = f2bf(x);
                }
            }
}

// ---------------------------------------------------------------------------
// Flash attention with analytic relative-position bias.
// Grid: (SEQ/64, B*H). Block: 128 threads = 4 waves; wave owns 16 query rows.
// Q,K: [B,H,S,D] bf16.  Vt: [B,H,D,S] bf16.  ctx out: [B,S,E] bf16.
// ---------------------------------------------------------------------------
__global__ __launch_bounds__(128)
void attn_wmma(const __bf16* __restrict__ Q, const __bf16* __restrict__ K,
               const __bf16* __restrict__ Vt, const int* __restrict__ pos_row,
               const int* __restrict__ pos_col, const float* __restrict__ rel_table,
               __bf16* __restrict__ ctx) {
    __shared__ __bf16 ptile[4][16][40];   // per-wave 16x32 P tile (+8 pad)

    const int lane = threadIdx.x & 31;
    const int wave = threadIdx.x >> 5;
    const int nn = lane & 15, hi = lane >> 4;
    const int bh = blockIdx.y;
    const int b = bh / HEADS, h = bh % HEADS;
    const int q0 = blockIdx.x * 64 + wave * 16;

    const __bf16* Qbh = Q + (size_t)bh * SEQ * HDIM;
    const __bf16* Kbh = K + (size_t)bh * SEQ * HDIM;
    const __bf16* Vbh = Vt + (size_t)bh * HDIM * SEQ;

    // Q A-fragments for d-chunks 0/1, kept in registers for whole kv loop
    AFrag aq[2];
#pragma unroll
    for (int c = 0; c < 2; ++c) {
        const __bf16* p = Qbh + (size_t)(q0 + nn) * HDIM + c * 32 + hi * 16;
        aq[c].h[0] = *(const v8bf*)p;
        aq[c].h[1] = *(const v8bf*)(p + 8);
    }

    const float t0 = rel_table[0 * HEADS + h], t1 = rel_table[1 * HEADS + h];
    const float t2 = rel_table[2 * HEADS + h], t3 = rel_table[3 * HEADS + h];

    int qpr[8], qpc[8];
#pragma unroll
    for (int r = 0; r < 8; ++r) {
        int m = q0 + r + 8 * hi;           // row this C-register corresponds to
        qpr[r] = pos_row[b * SEQ + m];
        qpc[r] = pos_col[b * SEQ + m];
    }

    float mi[8], li[8];
#pragma unroll
    for (int r = 0; r < 8; ++r) { mi[r] = -1e30f; li[r] = 0.f; }
    v8f o[4] = {};

    for (int kb = 0; kb < SEQ; kb += 32) {
        v8f s[2];
#pragma unroll
        for (int sub = 0; sub < 2; ++sub) {
            const int kv0 = kb + sub * 16;
            // K B-fragments: lane = key (kv0+nn); d = c*32 + hi*16 + {0..15}
            const __bf16* kp = Kbh + (size_t)(kv0 + nn) * HDIM + hi * 16;
            AFrag bk0, bk1;
            bk0.h[0] = *(const v8bf*)kp;        bk0.h[1] = *(const v8bf*)(kp + 8);
            bk1.h[0] = *(const v8bf*)(kp + 32); bk1.h[1] = *(const v8bf*)(kp + 40);
            v8f z = {};
            z = __builtin_amdgcn_wmma_f32_16x16x32_bf16(false, aq[0].v, false, bk0.v,
                                                        (short)0, z, false, false);
            z = __builtin_amdgcn_wmma_f32_16x16x32_bf16(false, aq[1].v, false, bk1.v,
                                                        (short)0, z, false, false);
            // analytic relative-position bias (same cell > same col > same row > other)
            const int kpr = pos_row[b * SEQ + kv0 + nn];
            const int kpc = pos_col[b * SEQ + kv0 + nn];
#pragma unroll
            for (int r = 0; r < 8; ++r) {
                bool er = (qpr[r] == kpr), ec = (qpc[r] == kpc);
                z[r] += er ? (ec ? t3 : t1) : (ec ? t2 : t0);
            }
            s[sub] = z;
        }

        // ---- online softmax over this 32-key block ----
        float mnew[8], alpha[8];
#pragma unroll
        for (int r = 0; r < 8; ++r) {
            float v = fmaxf(s[0][r], s[1][r]);
            v = fmaxf(v, __shfl_xor(v, 1));
            v = fmaxf(v, __shfl_xor(v, 2));
            v = fmaxf(v, __shfl_xor(v, 4));
            v = fmaxf(v, __shfl_xor(v, 8));
            mnew[r] = fmaxf(mi[r], v);
            alpha[r] = __expf(mi[r] - mnew[r]);
            mi[r] = mnew[r];
        }
#pragma unroll
        for (int r = 0; r < 8; ++r) {
            s[0][r] = __expf(s[0][r] - mnew[r]);
            s[1][r] = __expf(s[1][r] - mnew[r]);
            float t = s[0][r] + s[1][r];
            t += __shfl_xor(t, 1);
            t += __shfl_xor(t, 2);
            t += __shfl_xor(t, 4);
            t += __shfl_xor(t, 8);
            li[r] = li[r] * alpha[r] + t;
        }
#pragma unroll
        for (int nt = 0; nt < 4; ++nt)
#pragma unroll
            for (int r = 0; r < 8; ++r) o[nt][r] *= alpha[r];

        // ---- repack P: C layout -> A layout via LDS ----
#pragma unroll
        for (int r = 0; r < 8; ++r) {
            ptile[wave][r + 8 * hi][nn]      = f2bf(s[0][r]);
            ptile[wave][r + 8 * hi][nn + 16] = f2bf(s[1][r]);
        }
        __syncthreads();
        AFrag ap;
        const __bf16* rp = &ptile[wave][nn][hi * 16];
        ap.h[0] = *(const v8bf*)rp;
        ap.h[1] = *(const v8bf*)(rp + 8);
        __syncthreads();

        // ---- O += P @ V : B-fragment lane = col d (nt*16+nn); k = kv kb+hi*16+{0..15}
#pragma unroll
        for (int nt = 0; nt < 4; ++nt) {
            const __bf16* vp = Vbh + (size_t)(nt * 16 + nn) * SEQ + kb + hi * 16;
            AFrag bv;
            bv.h[0] = *(const v8bf*)vp;
            bv.h[1] = *(const v8bf*)(vp + 8);
            o[nt] = __builtin_amdgcn_wmma_f32_16x16x32_bf16(
                false, ap.v, false, bv.v, (short)0, o[nt], false, false);
        }
    }

    // normalize and write ctx as [B, S, E] bf16 for the output projection GEMM
#pragma unroll
    for (int r = 0; r < 8; ++r) li[r] = 1.f / li[r];
#pragma unroll
    for (int nt = 0; nt < 4; ++nt)
#pragma unroll
        for (int r = 0; r < 8; ++r) {
            int srow = q0 + r + 8 * hi;
            int col = h * HDIM + nt * 16 + nn;
            ctx[((size_t)b * SEQ + srow) * EMBED + col] = f2bf(o[nt][r] * li[r]);
        }
}

// ---------------------------------------------------------------------------
// Host-side orchestration
// ---------------------------------------------------------------------------
extern "C" void kernel_launch(void* const* d_in, const int* in_sizes, int n_in,
                              void* d_out, int out_size, void* d_ws, size_t ws_size,
                              hipStream_t stream) {
    const float* hid = (const float*)d_in[0];
    const int*   pr  = (const int*)d_in[1];
    const int*   pc  = (const int*)d_in[2];
    const float* qw  = (const float*)d_in[3];
    const float* qb  = (const float*)d_in[4];
    const float* kw  = (const float*)d_in[5];
    const float* kbs = (const float*)d_in[6];
    const float* vw  = (const float*)d_in[7];
    const float* vb  = (const float*)d_in[8];
    const float* ow  = (const float*)d_in[9];
    const float* ob  = (const float*)d_in[10];
    const float* rt  = (const float*)d_in[11];

    // workspace layout (bf16): hid | 4 weights | Q | K | Vt | ctx  (~35 MB total)
    char* ws = (char*)d_ws;
    const size_t ACT = (size_t)MROWS * EMBED * sizeof(__bf16);   // 6.29 MB
    const size_t WEI = (size_t)EMBED * EMBED * sizeof(__bf16);   // 1.18 MB
    __bf16* hidB = (__bf16*)ws;  ws += ACT;
    __bf16* qwB  = (__bf16*)ws;  ws += WEI;
    __bf16* kwB  = (__bf16*)ws;  ws += WEI;
    __bf16* vwB  = (__bf16*)ws;  ws += WEI;
    __bf16* owB  = (__bf16*)ws;  ws += WEI;
    __bf16* Qb   = (__bf16*)ws;  ws += ACT;
    __bf16* Kb   = (__bf16*)ws;  ws += ACT;
    __bf16* Vtb  = (__bf16*)ws;  ws += ACT;
    __bf16* Ctx  = (__bf16*)ws;  ws += ACT;

    cvt_f32_to_bf16<<<512, 256, 0, stream>>>(hid, hidB, MROWS * EMBED);
    cvt_f32_to_bf16<<<256, 256, 0, stream>>>(qw, qwB, EMBED * EMBED);
    cvt_f32_to_bf16<<<256, 256, 0, stream>>>(kw, kwB, EMBED * EMBED);
    cvt_f32_to_bf16<<<256, 256, 0, stream>>>(vw, vwB, EMBED * EMBED);
    cvt_f32_to_bf16<<<256, 256, 0, stream>>>(ow, owB, EMBED * EMBED);

    dim3 gg(MROWS / 64, EMBED / 128);   // 64 x 6 blocks
    gemm_bf16_wmma<<<gg, 256, 0, stream>>>(hidB, qwB, qb, SCALING, 1, nullptr, Qb);
    gemm_bf16_wmma<<<gg, 256, 0, stream>>>(hidB, kwB, kbs, 1.0f,   2, nullptr, Kb);
    gemm_bf16_wmma<<<gg, 256, 0, stream>>>(hidB, vwB, vb, 1.0f,    3, nullptr, Vtb);

    attn_wmma<<<dim3(SEQ / 64, BATCH * HEADS), 128, 0, stream>>>(
        Qb, Kb, Vtb, pr, pc, rt, Ctx);

    gemm_bf16_wmma<<<gg, 256, 0, stream>>>(Ctx, owB, ob, 1.0f, 0,
                                           (float*)d_out, nullptr);
}
